// LiquidTimeConstantBlock_60163901882481
// MI455X (gfx1250) — matrix operand order repaired and checked by reading, not compiled, not run on hardware
//
#include <hip/hip_runtime.h>
#include <hip/hip_bf16.h>

typedef __attribute__((ext_vector_type(16))) __bf16 v16bf;
typedef __attribute__((ext_vector_type(8)))  __bf16 v8bf;
typedef __attribute__((ext_vector_type(8)))  float  v8f;

#define B_SZ 64
#define T_SZ 512
#define D_SZ 1024
#define BT_SZ (B_SZ * T_SZ)   // 32768 rows
#define G_WGS 32               // persistent workgroups in the scan kernel

union ABu { v16bf v; v8bf h[2]; };

__device__ __forceinline__ float sigmoidf_(float x) { return 1.0f / (1.0f + __expf(-x)); }

// Async 16-byte global -> LDS copy (ASYNCcnt path, CDNA5).
__device__ __forceinline__ void async_g2l_b128(unsigned ldsoff, const void* gptr) {
  unsigned long long ga = (unsigned long long)(uintptr_t)gptr;
  asm volatile("global_load_async_to_lds_b128 %0, %1, off"
               :: "v"(ldsoff), "v"(ga) : "memory");
}
__device__ __forceinline__ void wait_asynccnt0() {
  asm volatile("s_wait_asynccnt 0x0" ::: "memory");
}

// ---------------------------------------------------------------------------
// adaptive_thresh = sigmoid(log_thresh): to workspace (for scan) and d_out tail
// ---------------------------------------------------------------------------
__global__ void k_thresh(const float* __restrict__ lt, float* __restrict__ thr_ws,
                         float* __restrict__ thr_out) {
  int i = threadIdx.x + blockIdx.x * blockDim.x;
  if (i < D_SZ) {
    float v = sigmoidf_(lt[i]);
    thr_ws[i]  = v;
    thr_out[i] = v;
  }
}

// ---------------------------------------------------------------------------
// f32 -> bf16 (8 elements / thread, vectorized)
// ---------------------------------------------------------------------------
__global__ void k_cvt_bf16(const float* __restrict__ src, __bf16* __restrict__ dst, int n8) {
  int i = threadIdx.x + blockIdx.x * blockDim.x;
  if (i >= n8) return;
  const float4* s = (const float4*)src + (size_t)i * 2;
  float4 a = s[0], b = s[1];
  v8bf o = { (__bf16)a.x, (__bf16)a.y, (__bf16)a.z, (__bf16)a.w,
             (__bf16)b.x, (__bf16)b.y, (__bf16)b.z, (__bf16)b.w };
  *((v8bf*)dst + i) = o;
}

// ---------------------------------------------------------------------------
// W (DxD f32, row-major) -> W^T (DxD bf16). 32x32 tiles via LDS.
// ---------------------------------------------------------------------------
__global__ void k_transpose_bf16(const float* __restrict__ src, __bf16* __restrict__ dst) {
  __shared__ float tile[32][33];
  int bx = blockIdx.x * 32;  // n dim
  int by = blockIdx.y * 32;  // k dim
  int tx = threadIdx.x, ty = threadIdx.y;   // block (32, 8)
  #pragma unroll
  for (int j = 0; j < 32; j += 8)
    tile[ty + j][tx] = src[(size_t)(by + ty + j) * D_SZ + bx + tx];
  __syncthreads();
  #pragma unroll
  for (int j = 0; j < 32; j += 8)
    dst[(size_t)(bx + ty + j) * D_SZ + by + tx] = (__bf16)tile[tx][ty + j];
}

// ---------------------------------------------------------------------------
// Big GEMM: out[M,D] = A[M,D(bf16)] * BT[D,D(bf16, pre-transposed)]^T + bias
// Block: 256 threads = 8 waves; block tile 128(M) x 64(N); wave: 16x64 (4 wmma acc).
// B tile (64 x 32k, 4KB) double-buffered in LDS, staged with
// global_load_async_to_lds_b128 (ASYNCcnt); all 4 B operand tiles are loaded
// into independent registers before the WMMA chain so DS waits can be partial.
// ---------------------------------------------------------------------------
__global__ void __launch_bounds__(256)
k_gemm_bf16(const __bf16* __restrict__ A, const __bf16* __restrict__ BT,
            const float* __restrict__ bias, float* __restrict__ out) {
  __shared__ __bf16 Bs[2][64 * 32];
  const int tid   = threadIdx.x;
  const int wave  = tid >> 5, lane = tid & 31;
  const int m0    = blockIdx.x * 128 + wave * 16;
  const int n0    = blockIdx.y * 64;
  const int lrow  = lane & 15;
  const int kselA = (lane & 16) ? 8 : 0;    // A: lanes>=16 hold K 8..15 / 24..31
  const int kselB = (lane & 16) ? 8 : 0;    // B: lanes>=16 hold K 8..15 / 24..31
  v8f acc[4] = {};
  const __bf16* arow = A + (size_t)(m0 + lrow) * D_SZ;

  // staging map: thread tid copies 16B: row tid>>2, k-chunk (tid&3)*8
  const int srow = tid >> 2, skc = (tid & 3) * 8;
  const __bf16* bsrc = &BT[(size_t)(n0 + srow) * D_SZ + skc];
  const unsigned lds0 = (unsigned)(uintptr_t)&Bs[0][srow * 32 + skc];
  const unsigned lds1 = (unsigned)(uintptr_t)&Bs[1][srow * 32 + skc];

  async_g2l_b128(lds0, bsrc);                       // stage tile k0=0 into buf 0
  for (int k0 = 0; k0 < D_SZ; k0 += 32) {
    const int buf = (k0 >> 5) & 1;
    ABu a;                                          // A loads overlap async wait
    a.h[0] = *(const v8bf*)&arow[k0 + kselA];
    a.h[1] = *(const v8bf*)&arow[k0 + kselA + 16];
    wait_asynccnt0();                               // my part of buf is in LDS
    __syncthreads();                                // everyone's part is in LDS
    if (k0 + 32 < D_SZ)
      async_g2l_b128(buf ? lds0 : lds1, bsrc + k0 + 32);  // stage next tile
    // issue all 8 ds_load_b128 up front into independent registers
    ABu b[4];
    #pragma unroll
    for (int tn = 0; tn < 4; ++tn) {
      const __bf16* bp = &Bs[buf][(tn * 16 + lrow) * 32 + kselB];
      b[tn].h[0] = *(const v8bf*)bp;
      b[tn].h[1] = *(const v8bf*)(bp + 16);
    }
    #pragma unroll
    for (int tn = 0; tn < 4; ++tn)
      acc[tn] = __builtin_amdgcn_wmma_f32_16x16x32_bf16(
          false, a.v, false, b[tn].v, (short)0, acc[tn], false, false);
  }
  const int mrow = (lane & 16) ? 8 : 0;  // D layout: lanes>=16 hold M=8..15
  #pragma unroll
  for (int tn = 0; tn < 4; ++tn) {
    int col = n0 + tn * 16 + lrow;
    float bsv = bias[col];
    #pragma unroll
    for (int r = 0; r < 8; ++r)
      out[(size_t)(m0 + mrow + r) * D_SZ + col] = acc[tn][r] + bsv;
  }
}

// ---------------------------------------------------------------------------
// Reset barrier counters (every launch -> graph-replay deterministic) and
// init tau state buffer 0 to bf16(1.0).
// ---------------------------------------------------------------------------
__global__ void k_init(unsigned* __restrict__ sync, __bf16* __restrict__ tau0) {
  int i = threadIdx.x + blockIdx.x * blockDim.x;
  if (i < 1024) sync[i] = 0u;
  if (i < B_SZ * D_SZ) tau0[i] = (__bf16)1.0f;
}

// ---------------------------------------------------------------------------
// Persistent scan kernel. 32 WGs x 256 threads (8 waves). WG g owns columns
// [g*32, g*32+32). Its 64KB slab of Wt_tau^T is async-staged into LDS once
// for all 512 steps. Wave w computes the 16x16 tile (rows (w&3)*16,
// cols (w>>2)*16) of tau_prev @ Wt_tau each step; the K-loop is software-
// pipelined (ping/pong operand registers, 64 K per iteration) so A (global)
// and B (LDS) loads overlap the WMMA accumulation chain. v state lives in
// registers for the whole scan. Grid-wide barrier per step via per-step
// atomic counters (release/acquire). xtau/mem/spikes are touch-once
// streams: prefetch early, non-temporal access.
// ---------------------------------------------------------------------------
__global__ void __launch_bounds__(256)
k_scan(const __bf16* __restrict__ WttT, const float* __restrict__ xtau,
       const float* __restrict__ mem, const float* __restrict__ thr,
       __bf16* __restrict__ tau_buf0, __bf16* __restrict__ tau_buf1,
       float* __restrict__ spikes, float* __restrict__ tau_out,
       float* __restrict__ v_out, unsigned* __restrict__ sync) {
  __shared__ __bf16 Ws[32 * 1024];   // 64 KB weight slab (32 cols x K=1024)
  const int wg    = blockIdx.x;
  const int tid   = threadIdx.x;
  const int wave  = tid >> 5, lane = tid & 31;
  const int mt    = wave & 3;        // row tile: batch rows mt*16..+16
  const int ct    = wave >> 2;       // col tile within the WG's 32 columns
  const int lrow  = lane & 15;
  const int kselA = (lane & 16) ? 8 : 0;
  const int kselB = (lane & 16) ? 8 : 0;
  const int mrow  = (lane & 16) ? 8 : 0;

  // Async preload of Wt_tau^T slab: rows wg*32 .. wg*32+31 (contiguous), 64 KB.
  for (int i = tid * 8; i < 32 * 1024; i += 256 * 8)
    async_g2l_b128((unsigned)(uintptr_t)&Ws[i], &WttT[(size_t)wg * 32 * 1024 + i]);
  wait_asynccnt0();
  __syncthreads();

  const int col = wg * 32 + ct * 16 + lrow;   // output column for this lane
  const __bf16* brow = &Ws[(ct * 16 + lrow) * 1024];
  const int arow_idx = mt * 16 + lrow;        // tau-state row read for A operand
  float vst[8] = {0, 0, 0, 0, 0, 0, 0, 0};
  const float thv = thr[col];
  float taunew[8];

  for (int t = 0; t < T_SZ; ++t) {
    const __bf16* taup = (t & 1) ? tau_buf1 : tau_buf0;
    __bf16*       taun = (t & 1) ? tau_buf0 : tau_buf1;

    // prefetch this step's xtau/mem elements; K-loop below hides their latency
    #pragma unroll
    for (int r = 0; r < 8; ++r) {
      int bidx = mt * 16 + mrow + r;
      int eoff = (bidx * T_SZ + t) * D_SZ + col;
      __builtin_prefetch(&xtau[eoff], 0, 1);
      __builtin_prefetch(&mem[eoff], 0, 1);
    }

    // 16x16 tile of tau_prev @ Wt_tau, software-pipelined over K
    v8f acc = {};
    const __bf16* arow = taup + (size_t)arow_idx * D_SZ;
    ABu a0, b0, a1, b1;
    a0.h[0] = *(const v8bf*)&arow[kselA];
    a0.h[1] = *(const v8bf*)&arow[kselA + 16];
    b0.h[0] = *(const v8bf*)&brow[kselB];
    b0.h[1] = *(const v8bf*)&brow[kselB + 16];
    #pragma unroll 2
    for (int k0 = 0; k0 < D_SZ; k0 += 64) {
      a1.h[0] = *(const v8bf*)&arow[k0 + 32 + kselA];
      a1.h[1] = *(const v8bf*)&arow[k0 + 32 + kselA + 16];
      b1.h[0] = *(const v8bf*)&brow[k0 + 32 + kselB];
      b1.h[1] = *(const v8bf*)&brow[k0 + 32 + kselB + 16];
      acc = __builtin_amdgcn_wmma_f32_16x16x32_bf16(
          false, a0.v, false, b0.v, (short)0, acc, false, false);
      if (k0 + 64 < D_SZ) {
        a0.h[0] = *(const v8bf*)&arow[k0 + 64 + kselA];
        a0.h[1] = *(const v8bf*)&arow[k0 + 64 + kselA + 16];
        b0.h[0] = *(const v8bf*)&brow[k0 + 64 + kselB];
        b0.h[1] = *(const v8bf*)&brow[k0 + 64 + kselB + 16];
      }
      acc = __builtin_amdgcn_wmma_f32_16x16x32_bf16(
          false, a1.v, false, b1.v, (short)0, acc, false, false);
    }

    // elementwise LTC update for the 16x16 tile
    #pragma unroll
    for (int r = 0; r < 8; ++r) {
      int bidx = mt * 16 + mrow + r;                       // batch index 0..63
      int eoff = (bidx * T_SZ + t) * D_SZ + col;           // (B,T,D) flat
      float tn = sigmoidf_(acc[r] + __builtin_nontemporal_load(&xtau[eoff]));
      taunew[r] = tn;
      float alpha = __expf(-1.0f / (tn + 1e-6f));
      float vv = alpha * vst[r] + (1.0f - alpha) * __builtin_nontemporal_load(&mem[eoff]);
      float s  = (vv >= thv) ? 1.0f : 0.0f;
      __builtin_nontemporal_store(s, &spikes[eoff]);
      vst[r] = vv * (1.0f - s);
      taun[bidx * D_SZ + col] = (__bf16)tn;
    }
    if (t == T_SZ - 1) {
      #pragma unroll
      for (int r = 0; r < 8; ++r) {
        int bidx = mt * 16 + mrow + r;
        tau_out[bidx * D_SZ + col] = taunew[r];
        v_out[bidx * D_SZ + col]   = vst[r];
      }
    }

    // grid-wide barrier so every WG sees the full tau state of step t
    __threadfence();
    __syncthreads();
    if (tid == 0) {
      __hip_atomic_fetch_add(&sync[t], 1u, __ATOMIC_RELEASE, __HIP_MEMORY_SCOPE_AGENT);
      while (__hip_atomic_load(&sync[t], __ATOMIC_ACQUIRE, __HIP_MEMORY_SCOPE_AGENT) < G_WGS)
        __builtin_amdgcn_s_sleep(1);
    }
    __syncthreads();
  }
}

// ---------------------------------------------------------------------------
extern "C" void kernel_launch(void* const* d_in, const int* in_sizes, int n_in,
                              void* d_out, int out_size, void* d_ws, size_t ws_size,
                              hipStream_t stream) {
  (void)in_sizes; (void)n_in; (void)out_size; (void)ws_size;
  const float* x  = (const float*)d_in[0];
  const float* Wt = (const float*)d_in[1];
  const float* bt = (const float*)d_in[2];
  const float* Wm = (const float*)d_in[3];
  const float* bm = (const float*)d_in[4];
  const float* lt = (const float*)d_in[5];

  float* out     = (float*)d_out;
  float* spikes  = out;                                    // B*T*D
  float* tau_out = out + (size_t)BT_SZ * D_SZ;             // B*D
  float* v_out   = tau_out + (size_t)B_SZ * D_SZ;          // B*D
  float* thr_out = v_out + (size_t)B_SZ * D_SZ;            // D

  char* ws = (char*)d_ws;
  size_t o = 0;
  __bf16* xbf  = (__bf16*)(ws + o); o += (size_t)BT_SZ * D_SZ * 2;  // 67 MB
  __bf16* WtxT = (__bf16*)(ws + o); o += (size_t)D_SZ * D_SZ * 2;   // 2 MB
  __bf16* WttT = (__bf16*)(ws + o); o += (size_t)D_SZ * D_SZ * 2;
  __bf16* WmT  = (__bf16*)(ws + o); o += (size_t)D_SZ * D_SZ * 2;
  float* xtau  = (float*)(ws + o);  o += (size_t)BT_SZ * D_SZ * 4;  // 134 MB
  float* mem   = (float*)(ws + o);  o += (size_t)BT_SZ * D_SZ * 4;  // 134 MB
  __bf16* tau0 = (__bf16*)(ws + o); o += (size_t)B_SZ * D_SZ * 2;
  __bf16* tau1 = (__bf16*)(ws + o); o += (size_t)B_SZ * D_SZ * 2;
  float* thr   = (float*)(ws + o);  o += 4096;
  unsigned* syn = (unsigned*)(ws + o); o += 4096;

  k_thresh<<<1, 1024, 0, stream>>>(lt, thr, thr_out);
  k_cvt_bf16<<<(BT_SZ * D_SZ / 8 + 255) / 256, 256, 0, stream>>>(x, xbf, BT_SZ * D_SZ / 8);

  dim3 tb(32, 8), tg(32, 32);
  k_transpose_bf16<<<tg, tb, 0, stream>>>(Wt, WtxT);                          // Wt[:D]  -> Wt_x^T
  k_transpose_bf16<<<tg, tb, 0, stream>>>(Wt + (size_t)D_SZ * D_SZ, WttT);    // Wt[D:]  -> Wt_tau^T
  k_transpose_bf16<<<tg, tb, 0, stream>>>(Wm, WmT);                           // Wm      -> Wm^T

  dim3 gg(BT_SZ / 128, D_SZ / 64);
  k_gemm_bf16<<<gg, 256, 0, stream>>>(xbf, WtxT, bt, xtau);
  k_gemm_bf16<<<gg, 256, 0, stream>>>(xbf, WmT,  bm, mem);

  k_init<<<(B_SZ * D_SZ + 255) / 256, 256, 0, stream>>>(syn, tau0);
  k_scan<<<G_WGS, 256, 0, stream>>>(WttT, xtau, mem, thr, tau0, tau1,
                                    spikes, tau_out, v_out, syn);
}
